// BiLSTMCRF_37263136260490
// MI455X (gfx1250) — compile-verified, matrix operand order checked by reading, not run on hardware
//
#include <hip/hip_runtime.h>
#include <hip/hip_bf16.h>

typedef __attribute__((ext_vector_type(16))) _Float16 v16h;
typedef __attribute__((ext_vector_type(8)))  float    v8f;

#define S_LEN 512
#define BATCH 64
#define EMB_D 512
#define HSZ   512      // per-direction hidden
#define G4    2048     // 4*H
#define TAGS_N 64
#define NEGV  -10000.0f

// ---- WMMA helpers -----------------------------------------------------------

__device__ __forceinline__ v8f wmma_f16(v16h a, v16h b, v8f c) {
  // D = A(16x32 f16) * B(32x16 f16) + C(16x16 f32)
  return __builtin_amdgcn_wmma_f32_16x16x32_f16(
      /*neg_a=*/false, a, /*neg_b=*/false, b,
      /*c_mod=*/(short)0, c, /*reuse_a=*/false, /*reuse_b=*/false);
}

// K-offset inside a 32-wide K block for 16-bit A/B fragments (ISA 7.12.2):
// lanes 0-15: VGPR0..3 -> K 0..7, VGPR4..7 -> K 16..23
// lanes16-31: VGPR0..3 -> K 8..15, VGPR4..7 -> K 24..31
__device__ __forceinline__ int frag_koff(int j, int lane) {
  return ((j & 4) ? 16 : 0) + ((lane & 16) ? 8 : 0) + ((j & 3) << 1);
}

// A fragment from row-major [M][K] f16 (global or LDS-generic pointer).
// Per lane this is two contiguous 16-byte segments -> 2x b128 loads.
__device__ __forceinline__ v16h load_a_rowmajor(const _Float16* base, int row,
                                                int ldk, int kbase, int lane) {
  v16h f;
  const _Float16* p = base + (size_t)row * ldk + kbase;
#pragma unroll
  for (int j = 0; j < 8; ++j) {
    int k = frag_koff(j, lane);
    f[2 * j]     = p[k];
    f[2 * j + 1] = p[k + 1];
  }
  return f;
}

// B fragment from prepacked layout: one aligned 32-byte load per lane.
__device__ __forceinline__ v16h load_b_packed(const _Float16* base, int blk,
                                              int lane) {
  return ((const v16h*)base)[(size_t)blk * 32 + lane];
}

// ---- Prep kernels -----------------------------------------------------------

// Pack fp32 weight src (N rows x K cols, row-major; row n = output column,
// col k = reduction index) into WMMA B-fragment-major f16 layout:
//   block (nt, kbi) -> 32 lanes x v16h, lane holds its exact fragment.
// Block index = nt * (K/32) + kbi  (contiguous along k for fixed n-tile).
__global__ void k_pack_b(const float* __restrict__ src,
                         _Float16* __restrict__ dst, int N, int K) {
  int t = blockIdx.x * blockDim.x + threadIdx.x;   // exact grid
  int lane = t & 31;
  int blk  = t >> 5;
  int kt   = K >> 5;                 // k-blocks
  int nt   = blk / kt;
  int kbi  = blk - nt * kt;
  int n    = nt * 16 + (lane & 15);
  v16h v;
#pragma unroll
  for (int j = 0; j < 8; ++j) {
    int k = kbi * 32 + frag_koff(j, lane);
    v[2 * j]     = (_Float16)src[(size_t)n * K + k];
    v[2 * j + 1] = (_Float16)src[(size_t)n * K + k + 1];
  }
  ((v16h*)dst)[(size_t)blk * 32 + lane] = v;
}

// xs16[(s*64+b)*512 + e] = emb[sentences[b*512+s]][e]  (time-major, f16)
__global__ void k_embed(const int* __restrict__ sent,
                        const float* __restrict__ emb,
                        _Float16* __restrict__ xs) {
  size_t i = (size_t)blockIdx.x * blockDim.x + threadIdx.x;  // exact grid
  int e = (int)(i & (EMB_D - 1));
  int m = (int)(i >> 9);            // s*64 + b
  int s = m >> 6, b = m & 63;
  int tok = sent[b * S_LEN + s];
  xs[i] = (_Float16)emb[(size_t)tok * EMB_D + e];
}

// ---- Input GEMM: pre = xs @ w_ih^T + (b_ih + b_hh), both directions ---------
// A: xs16 (32768 x 512) row-major, B: packed wih (N=2048, K=512), out f16.
// Each wave computes a 16x64 strip (4 n-tiles) so each A fragment feeds 4 WMMAs.

__global__ __launch_bounds__(256) void k_pre_gemm(
    const _Float16* __restrict__ xs,
    const _Float16* __restrict__ wih_f, const _Float16* __restrict__ wih_b,
    const float* __restrict__ bihf, const float* __restrict__ bhhf,
    const float* __restrict__ bihb, const float* __restrict__ bhhb,
    _Float16* __restrict__ pre_f, _Float16* __restrict__ pre_b) {
  const int STRIPS = (S_LEN * BATCH / 16) * (G4 / 64);  // 2048 * 32
  int gw   = blockIdx.x * 8 + (threadIdx.x >> 5);
  int lane = threadIdx.x & 31;
  int dir  = (gw >= STRIPS) ? 1 : 0;
  int strip = gw - dir * STRIPS;
  int mt  = strip >> 5;     // 0..2047
  int ntg = strip & 31;     // group of 4 n-tiles
  const _Float16* w  = dir ? wih_b : wih_f;
  const float* bi    = dir ? bihb : bihf;
  const float* bh    = dir ? bhhb : bhhf;
  _Float16* out      = dir ? pre_b : pre_f;

  int arow = mt * 16 + (lane & 15);
  int nt0  = ntg * 4;
  const int KT = EMB_D >> 5;        // 16 k-blocks
  v8f acc[4] = {v8f{}, v8f{}, v8f{}, v8f{}};
#pragma unroll 2
  for (int kbi = 0; kbi < KT; ++kbi) {
    v16h a = load_a_rowmajor(xs, arow, EMB_D, kbi * 32, lane);
#pragma unroll
    for (int t = 0; t < 4; ++t) {
      v16h b = load_b_packed(w, (nt0 + t) * KT + kbi, lane);
      acc[t] = wmma_f16(a, b, acc[t]);
    }
  }
  int mbase = mt * 16 + ((lane & 16) ? 8 : 0);
#pragma unroll
  for (int t = 0; t < 4; ++t) {
    int bcol = (nt0 + t) * 16 + (lane & 15);
    float bias = bi[bcol] + bh[bcol];
#pragma unroll
    for (int r = 0; r < 8; ++r)
      out[(size_t)(mbase + r) * G4 + bcol] = (_Float16)(acc[t][r] + bias);
  }
}

// ---- Persistent bidirectional LSTM recurrence -------------------------------
// grid = 2 blocks (block 0: forward, block 1: backward), 1024 threads each.
// h kept in LDS (64x512 f16 = 64KB). Per step:
//   phase A: gates_scratch = h @ w_hh^T  (WMMA, 32 waves; 4 n-tiles per
//            accumulator group so each LDS A fragment feeds 4 WMMAs)
//   phase B: gate nonlinearities + cell update, h -> LDS + hs output (f16)

__global__ __launch_bounds__(1024) void k_lstm(
    const _Float16* __restrict__ pre_f, const _Float16* __restrict__ pre_b,
    const _Float16* __restrict__ whhf,  const _Float16* __restrict__ whhb,
    _Float16* __restrict__ hsf, _Float16* __restrict__ hsb,
    float* __restrict__ gscr_f, float* __restrict__ gscr_b) {
  __shared__ _Float16 hbuf[BATCH * HSZ];  // 64 KB

  const int dir = blockIdx.x;
  const _Float16* pre = dir ? pre_b : pre_f;
  const _Float16* whh = dir ? whhb : whhf;
  _Float16* hs        = dir ? hsb : hsf;
  float* g            = dir ? gscr_b : gscr_f;

  const int tid = threadIdx.x;
  for (int i = tid; i < BATCH * HSZ; i += 1024) hbuf[i] = (_Float16)0.0f;
  float c_reg[32];
#pragma unroll
  for (int q = 0; q < 32; ++q) c_reg[q] = 0.0f;
  __syncthreads();

  const int wave = tid >> 5, lane = tid & 31;
  const int mt = wave & 3;              // 4 M-tiles (64 rows)
  const int ntbase = (wave >> 2) * 16;  // 8 wave-groups * 16 N-tiles = 128
  const int arow = mt * 16 + (lane & 15);
  const int mbase = mt * 16 + ((lane & 16) ? 8 : 0);
  const int KT = HSZ >> 5;              // 16 k-blocks

  for (int step = 0; step < S_LEN; ++step) {
    const int s = dir ? (S_LEN - 1 - step) : step;

    // ---- phase A: WMMA h(64x512) x whh^T(512x2048) -> gate scratch (f32)
    for (int tg = 0; tg < 4; ++tg) {    // 4 groups of 4 n-tiles
      const int nt0 = ntbase + tg * 4;
      v8f acc[4] = {v8f{}, v8f{}, v8f{}, v8f{}};
#pragma unroll 2
      for (int kbi = 0; kbi < KT; ++kbi) {
        v16h a = load_a_rowmajor(hbuf, arow, HSZ, kbi * 32, lane);
#pragma unroll
        for (int t = 0; t < 4; ++t) {
          v16h b = load_b_packed(whh, (nt0 + t) * KT + kbi, lane);
          acc[t] = wmma_f16(a, b, acc[t]);
        }
      }
#pragma unroll
      for (int t = 0; t < 4; ++t) {
        int bcol = (nt0 + t) * 16 + (lane & 15);
#pragma unroll
        for (int r = 0; r < 8; ++r)
          g[(size_t)(mbase + r) * G4 + bcol] = acc[t][r];
      }
    }
    __threadfence();
    __syncthreads();

    // ---- phase B: gates -> (c, h)
    const size_t prow = (size_t)s * BATCH * G4;
#pragma unroll
    for (int q = 0; q < 32; ++q) {
      int p = tid + q * 1024;           // 0..32767
      int b = p >> 9, j = p & 511;
      size_t pb = prow + (size_t)b * G4;
      size_t gb = (size_t)b * G4;
      float gi = (float)pre[pb + j]            + g[gb + j];
      float gf = (float)pre[pb + HSZ + j]      + g[gb + HSZ + j];
      float gg = (float)pre[pb + 2 * HSZ + j]  + g[gb + 2 * HSZ + j];
      float go = (float)pre[pb + 3 * HSZ + j]  + g[gb + 3 * HSZ + j];
      float si = 1.0f / (1.0f + __expf(-gi));
      float sf = 1.0f / (1.0f + __expf(-gf));
      float so = 1.0f / (1.0f + __expf(-go));
      float cc = sf * c_reg[q] + si * tanhf(gg);
      float hh = so * tanhf(cc);
      c_reg[q] = cc;
      _Float16 h16 = (_Float16)hh;
      hbuf[b * HSZ + j] = h16;
      hs[((size_t)s * BATCH + b) * HSZ + j] = h16;
    }
    __syncthreads();
  }
}

// ---- Output projection: emissions = [h_f ; h_b] @ w_out^T + b_out -----------
// A: hsf/hsb (32768 x 512 each), B: packed wout (N=64, K=1024), out f32 (B,S,T)
// One wave per m-tile handles all 4 n-tiles: each A fragment feeds 4 WMMAs.

__global__ __launch_bounds__(256) void k_proj(
    const _Float16* __restrict__ hsf, const _Float16* __restrict__ hsb,
    const _Float16* __restrict__ wout, const float* __restrict__ bout,
    float* __restrict__ em) {
  int mt   = blockIdx.x * 8 + (threadIdx.x >> 5);  // 2048 m-tiles, exact
  int lane = threadIdx.x & 31;
  int arow = mt * 16 + (lane & 15);
  const int KT = (2 * HSZ) >> 5;     // 32 k-blocks total (16 fwd + 16 bwd)
  v8f acc[4] = {v8f{}, v8f{}, v8f{}, v8f{}};
#pragma unroll 2
  for (int kbi = 0; kbi < 16; ++kbi) {
    v16h a = load_a_rowmajor(hsf, arow, HSZ, kbi * 32, lane);
#pragma unroll
    for (int t = 0; t < 4; ++t) {
      v16h b = load_b_packed(wout, t * KT + kbi, lane);
      acc[t] = wmma_f16(a, b, acc[t]);
    }
  }
#pragma unroll 2
  for (int kbi = 0; kbi < 16; ++kbi) {
    v16h a = load_a_rowmajor(hsb, arow, HSZ, kbi * 32, lane);
#pragma unroll
    for (int t = 0; t < 4; ++t) {
      v16h b = load_b_packed(wout, t * KT + 16 + kbi, lane);
      acc[t] = wmma_f16(a, b, acc[t]);
    }
  }
  int mbase = mt * 16 + ((lane & 16) ? 8 : 0);
#pragma unroll
  for (int t = 0; t < 4; ++t) {
    int bcol = t * 16 + (lane & 15);
    float bias = bout[bcol];
#pragma unroll
    for (int r = 0; r < 8; ++r) {
      int m = mbase + r;               // m = s*64 + b
      int s = m >> 6, b = m & 63;
      em[((size_t)b * S_LEN + s) * TAGS_N + bcol] = acc[t][r] + bias;
    }
  }
}

// ---- CRF: forward (log-partition) + gold score per batch element ------------

__global__ __launch_bounds__(64) void k_crf(
    const float* __restrict__ em, const int* __restrict__ tags,
    const float* __restrict__ trans, float* __restrict__ res) {
  __shared__ float tl[TAGS_N * TAGS_N];   // 16 KB
  __shared__ float abuf[2][TAGS_N];
  __shared__ float red[TAGS_N];

  const int b = blockIdx.x, j = threadIdx.x;
  for (int i = j; i < TAGS_N * TAGS_N; i += TAGS_N) tl[i] = trans[i];
  abuf[0][j] = (j == 0) ? 0.0f : NEGV;
  __syncthreads();

  const float* emB = em + (size_t)b * S_LEN * TAGS_N;
  int cur = 0;
  for (int s = 0; s < S_LEN; ++s) {
    float e = emB[s * TAGS_N + j];
    const float* trow = &tl[j * TAGS_N];
    float mx = -3.0e38f;
#pragma unroll 8
    for (int k = 0; k < TAGS_N; ++k) mx = fmaxf(mx, abuf[cur][k] + trow[k]);
    float ss = 0.0f;
#pragma unroll 8
    for (int k = 0; k < TAGS_N; ++k) ss += __expf(abuf[cur][k] + trow[k] - mx);
    abuf[cur ^ 1][j] = e + mx + __logf(ss);
    __syncthreads();
    cur ^= 1;
  }

  // gold score contribution (64-way parallel over timesteps)
  float part = 0.0f;
  for (int s = j; s < S_LEN; s += TAGS_N) {
    int tg = tags[b * S_LEN + s];
    int pv = (s == 0) ? 0 : tags[b * S_LEN + s - 1];
    part += emB[s * TAGS_N + tg] + tl[tg * TAGS_N + pv];
  }
  red[j] = part;
  __syncthreads();

  if (j == 0) {
    float mx = abuf[cur][0];
    for (int k = 1; k < TAGS_N; ++k) mx = fmaxf(mx, abuf[cur][k]);
    float ss = 0.0f;
    for (int k = 0; k < TAGS_N; ++k) ss += __expf(abuf[cur][k] - mx);
    float fwd = mx + __logf(ss);
    float gold = 0.0f;
    for (int k = 0; k < TAGS_N; ++k) gold += red[k];
    res[b] = fwd - gold;
  }
}

__global__ void k_mean(const float* __restrict__ v, float* __restrict__ out) {
  if (threadIdx.x == 0) {
    float s = 0.0f;
    for (int i = 0; i < BATCH; ++i) s += v[i];
    out[0] = s / (float)BATCH;
  }
}

// ---- Host launch ------------------------------------------------------------

extern "C" void kernel_launch(void* const* d_in, const int* in_sizes, int n_in,
                              void* d_out, int out_size, void* d_ws,
                              size_t ws_size, hipStream_t stream) {
  const int*   sentences = (const int*)d_in[0];
  const int*   tags      = (const int*)d_in[1];
  // d_in[2] = lengths (unused; all == S)
  const float* emb   = (const float*)d_in[3];
  const float* wihf  = (const float*)d_in[4];
  const float* whhf  = (const float*)d_in[5];
  const float* bihf  = (const float*)d_in[6];
  const float* bhhf  = (const float*)d_in[7];
  const float* wihb  = (const float*)d_in[8];
  const float* whhb  = (const float*)d_in[9];
  const float* bihb  = (const float*)d_in[10];
  const float* bhhb  = (const float*)d_in[11];
  const float* wout  = (const float*)d_in[12];
  const float* bout  = (const float*)d_in[13];
  const float* trans = (const float*)d_in[14];
  float* out = (float*)d_out;

  char* ws = (char*)d_ws;
  size_t off = 0;
  auto alloc = [&](size_t bytes) -> void* {
    void* p = ws + off;
    off = (off + bytes + 255) & ~(size_t)255;
    return p;
  };
  const size_t MROWS = (size_t)S_LEN * BATCH;  // 32768
  _Float16* xs16    = (_Float16*)alloc(MROWS * EMB_D * 2);          // 32 MB
  _Float16* wihf16  = (_Float16*)alloc((size_t)EMB_D * G4 * 2);     //  2 MB
  _Float16* wihb16  = (_Float16*)alloc((size_t)EMB_D * G4 * 2);
  _Float16* whhf16  = (_Float16*)alloc((size_t)HSZ * G4 * 2);
  _Float16* whhb16  = (_Float16*)alloc((size_t)HSZ * G4 * 2);
  _Float16* wout16  = (_Float16*)alloc((size_t)(2 * HSZ) * TAGS_N * 2);
  _Float16* pre_f   = (_Float16*)alloc(MROWS * G4 * 2);             // 128 MB
  _Float16* pre_b   = (_Float16*)alloc(MROWS * G4 * 2);             // 128 MB
  _Float16* hsf     = (_Float16*)alloc(MROWS * HSZ * 2);            // 32 MB
  _Float16* hsb     = (_Float16*)alloc(MROWS * HSZ * 2);            // 32 MB
  float*    emis    = (float*)alloc(MROWS * TAGS_N * 4);            //  8 MB
  float*    gscr_f  = (float*)alloc((size_t)BATCH * G4 * 4);        // 512 KB
  float*    gscr_b  = (float*)alloc((size_t)BATCH * G4 * 4);
  float*    resB    = (float*)alloc(BATCH * 4);

  // 1) weight convert + pack into WMMA B-fragment layout (f16)
  //    threads = (N/16)*(K/32)*32
  k_pack_b<<<(G4 / 16) * (EMB_D / 32) * 32 / 256, 256, 0, stream>>>(
      wihf, wihf16, G4, EMB_D);
  k_pack_b<<<(G4 / 16) * (EMB_D / 32) * 32 / 256, 256, 0, stream>>>(
      wihb, wihb16, G4, EMB_D);
  k_pack_b<<<(G4 / 16) * (HSZ / 32) * 32 / 256, 256, 0, stream>>>(
      whhf, whhf16, G4, HSZ);
  k_pack_b<<<(G4 / 16) * (HSZ / 32) * 32 / 256, 256, 0, stream>>>(
      whhb, whhb16, G4, HSZ);
  k_pack_b<<<(TAGS_N / 16) * (2 * HSZ / 32) * 32 / 256, 256, 0, stream>>>(
      wout, wout16, TAGS_N, 2 * HSZ);

  // 2) embedding gather (time-major, f16)
  k_embed<<<(int)(MROWS * EMB_D / 256), 256, 0, stream>>>(sentences, emb, xs16);

  // 3) input GEMMs, both directions (WMMA, 4 n-tiles per wave)
  {
    int strips2 = 2 * (int)(MROWS / 16) * (G4 / 64);  // 131072 waves
    k_pre_gemm<<<strips2 / 8, 256, 0, stream>>>(xs16, wihf16, wihb16, bihf,
                                                bhhf, bihb, bhhb, pre_f, pre_b);
  }

  // 4) persistent bidirectional LSTM recurrence (WMMA per step)
  k_lstm<<<2, 1024, 0, stream>>>(pre_f, pre_b, whhf16, whhb16, hsf, hsb,
                                 gscr_f, gscr_b);

  // 5) output projection (WMMA, one wave per m-tile covers all 4 n-tiles)
  k_proj<<<(int)(MROWS / 16) / 8, 256, 0, stream>>>(hsf, hsb, wout16, bout,
                                                    emis);

  // 6) CRF forward + gold, then mean
  k_crf<<<BATCH, TAGS_N, 0, stream>>>(emis, tags, trans, resB);
  k_mean<<<1, 64, 0, stream>>>(resB, out);
  (void)in_sizes; (void)n_in; (void)out_size; (void)ws_size;
}
